// Encoder_26706106647176
// MI455X (gfx1250) — matrix operand order repaired and checked by reading, not compiled, not run on hardware
//
#include <hip/hip_runtime.h>
#include <hip/hip_bf16.h>
#include <math.h>

typedef __attribute__((ext_vector_type(16))) __bf16 v16bf;
typedef __attribute__((ext_vector_type(8)))  __bf16 v8bf;
typedef __attribute__((ext_vector_type(8)))  float  v8f;

#define MODE_NONE      0
#define MODE_PROP      1
#define MODE_PROP_RELU 2

// ---------------------------------------------------------------------------
// Tiled bf16 WMMA GEMM: C = A(MxK) * B(KxN), f32 accumulate.
// REQUIRES: M % 64 == 0, N % 64 == 0, K % 32 == 0 (operands padded with zeros).
// Stores are guarded by Nst (logical column count, Nst <= N).
// Block = 128 threads (4 waves), tile 64x64, K-step 32, software-pipelined:
// next slab is staged into registers while WMMAs consume the LDS slab.
// ---------------------------------------------------------------------------
__global__ __launch_bounds__(128) void gemm_bf16(
    const __bf16* __restrict__ A, int lda,
    const __bf16* __restrict__ B, int ldb,
    float* __restrict__ Cf, __bf16* __restrict__ Cb, int ldc,
    int M, int N, int K, int Nst, int mode,
    const float* __restrict__ l1, const float* __restrict__ l2,
    const float* __restrict__ D, int ldd)
{
    __shared__ __bf16 lsA[64][40];   // [row][k], pitch 40 keeps 16B alignment + breaks conflicts
    __shared__ __bf16 lsBT[64][40];  // [n][k]  (B transposed)

    const int tid  = threadIdx.x;
    const int lane = tid & 31;
    const int wave = tid >> 5;
    const int l16  = lane & 15;
    const int lh   = lane >> 4;           // lane half: 0 or 1

    const int row0 = blockIdx.y * 64;
    const int col0 = blockIdx.x * 64;
    const int wm   = (wave >> 1) * 32;
    const int wn   = (wave & 1)  * 32;

    // staging index maps (computed once)
    const int idxA0 = tid;            // A: 64 rows x 4 octs
    const int idxA1 = tid + 128;
    const int arr0 = idxA0 >> 2, aco0 = (idxA0 & 3) << 3;
    const int arr1 = idxA1 >> 2, aco1 = (idxA1 & 3) << 3;
    const int idxB0 = tid;            // B: 32 k x 8 n-octs
    const int idxB1 = tid + 128;
    const int bkk0 = idxB0 >> 3, bqn0 = (idxB0 & 7) << 3;
    const int bkk1 = idxB1 >> 3, bqn1 = (idxB1 & 7) << 3;

    const __bf16* Arow0 = A + (size_t)(row0 + arr0) * lda;
    const __bf16* Arow1 = A + (size_t)(row0 + arr1) * lda;
    const __bf16* Bp0   = B + (size_t)bkk0 * ldb + (col0 + bqn0);
    const __bf16* Bp1   = B + (size_t)bkk1 * ldb + (col0 + bqn1);
    const size_t  ldbK  = (size_t)32 * ldb;

    v8f acc00 = {}, acc01 = {}, acc10 = {}, acc11 = {};
    v8bf ra0, ra1, rb0, rb1;

    // prologue: load slab 0 into regs, commit to LDS
    ra0 = *(const v8bf*)(Arow0 + aco0);
    ra1 = *(const v8bf*)(Arow1 + aco1);
    rb0 = *(const v8bf*)(Bp0);
    rb1 = *(const v8bf*)(Bp1);
    *(v8bf*)&lsA[arr0][aco0] = ra0;
    *(v8bf*)&lsA[arr1][aco1] = ra1;
    #pragma unroll
    for (int j = 0; j < 8; ++j) lsBT[bqn0 + j][bkk0] = rb0[j];
    #pragma unroll
    for (int j = 0; j < 8; ++j) lsBT[bqn1 + j][bkk1] = rb1[j];
    __syncthreads();

    for (int k0 = 0; k0 < K; k0 += 32) {
        const bool more = (k0 + 32) < K;
        if (more) {
            // stage next slab into registers (overlaps with WMMA below)
            ra0 = *(const v8bf*)(Arow0 + (k0 + 32) + aco0);
            ra1 = *(const v8bf*)(Arow1 + (k0 + 32) + aco1);
            rb0 = *(const v8bf*)(Bp0 + (size_t)(k0 + 32) * ldb);
            rb1 = *(const v8bf*)(Bp1 + (size_t)(k0 + 32) * ldb);
            if (k0 + 64 < K) __builtin_prefetch(Arow0 + (k0 + 64) + aco0, 0, 0);
        }

        // ---- fragments (lane layouts per CDNA5 ISA 7.12.2) ----
        v16bf a0, a1, b0, b1;
        {
            const __bf16* pa0 = &lsA[wm + l16][lh * 8];
            const __bf16* pa1 = &lsA[wm + 16 + l16][lh * 8];
            v8bf alo0 = *(const v8bf*)(pa0);
            v8bf ahi0 = *(const v8bf*)(pa0 + 16);
            v8bf alo1 = *(const v8bf*)(pa1);
            v8bf ahi1 = *(const v8bf*)(pa1 + 16);
            const __bf16* pb0 = &lsBT[wn + l16][lh * 8];
            const __bf16* pb1 = &lsBT[wn + 16 + l16][lh * 8];
            v8bf blo0 = *(const v8bf*)(pb0);
            v8bf bhi0 = *(const v8bf*)(pb0 + 16);
            v8bf blo1 = *(const v8bf*)(pb1);
            v8bf bhi1 = *(const v8bf*)(pb1 + 16);
            #pragma unroll
            for (int i = 0; i < 8; ++i) {
                a0[i] = alo0[i]; a0[i + 8] = ahi0[i];
                a1[i] = alo1[i]; a1[i + 8] = ahi1[i];
                b0[i] = blo0[i]; b0[i + 8] = bhi0[i];
                b1[i] = blo1[i]; b1[i + 8] = bhi1[i];
            }
        }
        acc00 = __builtin_amdgcn_wmma_f32_16x16x32_bf16(false, a0, false, b0, (short)0, acc00, false, false);
        acc01 = __builtin_amdgcn_wmma_f32_16x16x32_bf16(false, a0, false, b1, (short)0, acc01, false, false);
        acc10 = __builtin_amdgcn_wmma_f32_16x16x32_bf16(false, a1, false, b0, (short)0, acc10, false, false);
        acc11 = __builtin_amdgcn_wmma_f32_16x16x32_bf16(false, a1, false, b1, (short)0, acc11, false, false);

        if (more) {
            __syncthreads();   // all frag reads done before overwriting LDS
            *(v8bf*)&lsA[arr0][aco0] = ra0;
            *(v8bf*)&lsA[arr1][aco1] = ra1;
            #pragma unroll
            for (int j = 0; j < 8; ++j) lsBT[bqn0 + j][bkk0] = rb0[j];
            #pragma unroll
            for (int j = 0; j < 8; ++j) lsBT[bqn1 + j][bkk1] = rb1[j];
            __syncthreads();
        }
    }

    // ---- epilogue: lane holds C[row0+wm+sm*16+8*lh+r][col0+wn+sn*16+l16] ----
    #pragma unroll
    for (int sm = 0; sm < 2; ++sm) {
        #pragma unroll
        for (int sn = 0; sn < 2; ++sn) {
            v8f a = (sm == 0) ? (sn == 0 ? acc00 : acc01)
                              : (sn == 0 ? acc10 : acc11);
            int gc = col0 + wn + sn * 16 + l16;
            if (gc >= Nst) continue;
            #pragma unroll
            for (int r = 0; r < 8; ++r) {
                int gr = row0 + wm + sm * 16 + lh * 8 + r;
                float v = a[r];
                if (mode == MODE_PROP || mode == MODE_PROP_RELU) {
                    v = l1[gr] * D[(size_t)gr * ldd + gc] + l2[gr] * v;
                    if (mode == MODE_PROP_RELU) v = v > 0.f ? v : 0.f;
                }
                if (Cf) Cf[(size_t)gr * ldc + gc] = v;
                if (Cb) Cb[(size_t)gr * ldc + gc] = (__bf16)v;
            }
        }
    }
}

// ---------------------------------------------------------------------------
// Small helper kernels
// ---------------------------------------------------------------------------

// f32 (r x c) -> bf16 (rp x cp), zero-padded
__global__ void conv_pad_k(const float* __restrict__ s, __bf16* __restrict__ d,
                           int r, int c, int rp, int cp) {
    long n  = (long)rp * cp;
    long i  = (long)blockIdx.x * blockDim.x + threadIdx.x;
    long st = (long)gridDim.x * blockDim.x;
    for (; i < n; i += st) {
        int rr = (int)(i / cp), cc = (int)(i % cp);
        d[i] = (rr < r && cc < c) ? (__bf16)s[(size_t)rr * c + cc] : (__bf16)0.0f;
    }
}

__global__ void relu_conv_k(const float* __restrict__ z, float* __restrict__ ef,
                            __bf16* __restrict__ eb, long n) {
    long i  = (long)blockIdx.x * blockDim.x + threadIdx.x;
    long st = (long)gridDim.x * blockDim.x;
    for (; i < n; i += st) {
        float v = z[i];
        v = v > 0.f ? v : 0.f;
        ef[i] = v;
        eb[i] = (__bf16)v;
    }
}

__global__ __launch_bounds__(256) void rowsum_k(const float* __restrict__ A,
                                                float* __restrict__ out, int cols) {
    __shared__ float red[256];
    int row = blockIdx.x;
    const float* p = A + (size_t)row * cols;
    float s = 0.f;
    for (int j = threadIdx.x; j < cols; j += 256) s += p[j];
    red[threadIdx.x] = s; __syncthreads();
    for (int o = 128; o > 0; o >>= 1) {
        if (threadIdx.x < o) red[threadIdx.x] += red[threadIdx.x + o];
        __syncthreads();
    }
    if (threadIdx.x == 0) out[row] = red[0];
}

// g = sigmoid(normalize(vsum / rowsum)), in place, cols == 64 == blockDim.x
__global__ __launch_bounds__(64) void readout_k(float* __restrict__ g,
                                                const float* __restrict__ rs, int cols) {
    __shared__ float red[64];
    int row = blockIdx.x;
    float v = g[(size_t)row * cols + threadIdx.x] / rs[row];
    red[threadIdx.x] = v * v; __syncthreads();
    for (int o = 32; o > 0; o >>= 1) {
        if (threadIdx.x < o) red[threadIdx.x] += red[threadIdx.x + o];
        __syncthreads();
    }
    float nrm = fmaxf(sqrtf(red[0]), 1e-12f);
    v /= nrm;
    g[(size_t)row * cols + threadIdx.x] = 1.f / (1.f + expf(-v));
}

// Bilinear disc: out[row] = {dot(pl_row, W@g_row)+db, dot(mi_row, W@g_row)+db}
__global__ __launch_bounds__(64) void disc_k(const float* __restrict__ g,
                                             const float* __restrict__ pl,
                                             const float* __restrict__ mi,
                                             const float* __restrict__ W,
                                             const float* __restrict__ db,
                                             float* __restrict__ out) {
    __shared__ float Ws[64 * 64];
    __shared__ float gs[64];
    __shared__ float r1[64], r2[64];
    int row = blockIdx.x;
    int d = threadIdx.x;
    for (int i = d; i < 64 * 64; i += 64) Ws[i] = W[i];
    gs[d] = g[(size_t)row * 64 + d];
    __syncthreads();
    float u = 0.f;
    #pragma unroll 8
    for (int e = 0; e < 64; ++e) u += Ws[d * 64 + e] * gs[e];
    r1[d] = pl[(size_t)row * 64 + d] * u;
    r2[d] = mi[(size_t)row * 64 + d] * u;
    __syncthreads();
    for (int o = 32; o > 0; o >>= 1) {
        if (d < o) { r1[d] += r1[d + o]; r2[d] += r2[d + o]; }
        __syncthreads();
    }
    if (d == 0) {
        out[(size_t)row * 2 + 0] = r1[0] + db[0];
        out[(size_t)row * 2 + 1] = r2[0] + db[0];
    }
}

__global__ void cluster_k(const float* __restrict__ z, const float* __restrict__ cw,
                          const float* __restrict__ cb, float* __restrict__ out, int n) {
    int i = blockIdx.x * blockDim.x + threadIdx.x;
    if (i >= n * 10) return;
    int row = i / 10, j = i % 10;
    const float* zr = z + (size_t)row * 64;
    const float* wr = cw + (size_t)j * 64;
    float s = cb[j];
    #pragma unroll 8
    for (int e = 0; e < 64; ++e) s += zr[e] * wr[e];
    out[i] = s;
}

__device__ float dino_term(const float* te, const float* st, const float* cen) {
    float ms = -1e30f, mt = -1e30f;
    for (int j = 0; j < 64; ++j) {
        ms = fmaxf(ms, st[j] / 0.9f);
        mt = fmaxf(mt, (te[j] - cen[j]) / 0.06f);
    }
    float ds = 0.f, dt = 0.f;
    for (int j = 0; j < 64; ++j) {
        ds += expf(st[j] / 0.9f - ms);
        dt += expf((te[j] - cen[j]) / 0.06f - mt);
    }
    float s = 0.f;
    for (int j = 0; j < 64; ++j) {
        float sp = expf(st[j] / 0.9f - ms) / ds;
        float tp = expf((te[j] - cen[j]) / 0.06f - mt) / dt;
        s += tp * logf(sp + 1e-20f);
    }
    return -s;
}

__global__ void dino_k(const float* __restrict__ z, const float* __restrict__ zm,
                       const float* __restrict__ cen, float* __restrict__ loss, int n) {
    int row = blockIdx.x * blockDim.x + threadIdx.x;
    if (row >= n) return;
    const float* zr = z + (size_t)row * 64;
    const float* mr = zm + (size_t)row * 64;
    float t = dino_term(zr, mr, cen) + dino_term(mr, zr, cen);
    atomicAdd(loss, t * (0.5f / (float)n));
}

// ---------------------------------------------------------------------------
extern "C" void kernel_launch(void* const* d_in, const int* in_sizes, int n_in,
                              void* d_out, int out_size, void* d_ws, size_t ws_size,
                              hipStream_t stream) {
    const int N = 4096, FIN = 3000, HID = 512, OUT = 64, NC = 10;
    const int FINP = 3008;  // FIN padded to multiple of 64 (and 32 for K)
    const float* feat   = (const float*)d_in[0];
    const float* feat_a = (const float*)d_in[1];
    const float* feat_m = (const float*)d_in[2];
    const float* adj    = (const float*)d_in[3];
    /* d_in[4] identify_matrix: folded into prop (identity) */
    const float* gn  = (const float*)d_in[5];
    const float* w21 = (const float*)d_in[6];
    const float* w22 = (const float*)d_in[7];
    const float* w2  = (const float*)d_in[8];
    const float* l1  = (const float*)d_in[9];
    const float* l2  = (const float*)d_in[10];
    const float* cw  = (const float*)d_in[11];
    const float* cb  = (const float*)d_in[12];
    const float* dw  = (const float*)d_in[13];
    const float* db  = (const float*)d_in[14];
    const float* cen = (const float*)d_in[15];
    float* out = (float*)d_out;

    // output offsets (floats, concatenated in return order)
    const size_t off_h    = (size_t)N * OUT;
    const size_t off_ret  = off_h + (size_t)N * FIN;
    const size_t off_reta = off_ret + (size_t)N * 2;
    const size_t off_retm = off_reta + (size_t)N * 2;
    const size_t off_c    = off_retm + (size_t)N * 2;
    const size_t off_cm   = off_c + (size_t)N * NC;
    const size_t off_loss = off_cm + (size_t)N * NC;

    // bump allocator over workspace
    char* base = (char*)d_ws;
    size_t cur = 0;
    auto alloc = [&](size_t bytes) -> void* {
        void* p = base + cur;
        cur = (cur + bytes + 255) & ~(size_t)255;
        return p;
    };
    __bf16* adjb = (__bf16*)alloc((size_t)N * N * 2);
    __bf16* gnb  = (__bf16*)alloc((size_t)N * N * 2);
    __bf16* xb   = (__bf16*)alloc((size_t)N * FINP * 2);   // reused as yb after branches
    __bf16* w21b = (__bf16*)alloc((size_t)FINP * HID * 2);
    __bf16* w22b = (__bf16*)alloc((size_t)HID * OUT * 2);
    __bf16* w2b  = (__bf16*)alloc((size_t)OUT * FINP * 2);
    float*  t1f  = (float*) alloc((size_t)N * HID * 4);
    __bf16* t1b  = (__bf16*)alloc((size_t)N * HID * 2);
    __bf16* z1b  = (__bf16*)alloc((size_t)N * HID * 2);
    float*  t2f  = (float*) alloc((size_t)N * OUT * 4);
    __bf16* t2b  = (__bf16*)alloc((size_t)N * OUT * 2);
    __bf16* zb0  = (__bf16*)alloc((size_t)N * OUT * 2);
    float*  zaf  = (float*) alloc((size_t)N * OUT * 4);
    float*  zmf  = (float*) alloc((size_t)N * OUT * 4);
    float*  embf[3]; __bf16* embb[3]; float* gbuf[3];
    for (int i = 0; i < 3; ++i) {
        embf[i] = (float*) alloc((size_t)N * OUT * 4);
        embb[i] = (__bf16*)alloc((size_t)N * OUT * 2);
        gbuf[i] = (float*) alloc((size_t)N * OUT * 4);
    }
    float* gnrs = (float*)alloc((size_t)N * 4);
    (void)ws_size; (void)in_sizes; (void)n_in; (void)out_size;

    auto conv = [&](const float* s, __bf16* d, int r, int c, int rp, int cp) {
        long n = (long)rp * cp;
        long b = (n + 255) / 256; if (b > 4096) b = 4096;
        conv_pad_k<<<dim3((unsigned)b), dim3(256), 0, stream>>>(s, d, r, c, rp, cp);
    };
    auto gemm = [&](const __bf16* A, int lda, const __bf16* B, int ldb,
                    float* Cf, __bf16* Cb, int ldc, int M, int Nn, int K, int Nst,
                    int mode, const float* L1, const float* L2, const float* D, int ldd) {
        dim3 g((Nn + 63) / 64, (M + 63) / 64);
        gemm_bf16<<<g, dim3(128), 0, stream>>>(A, lda, B, ldb, Cf, Cb, ldc,
                                               M, Nn, K, Nst, mode, L1, L2, D, ldd);
    };

    // one-time conversions (zero-padded where needed)
    conv(adj, adjb, N, N, N, N);
    conv(gn,  gnb,  N, N, N, N);
    conv(w21, w21b, FIN, HID, FINP, HID);
    conv(w22, w22b, HID, OUT, HID, OUT);
    conv(w2,  w2b,  OUT, FIN, OUT, FINP);
    rowsum_k<<<dim3(N), dim3(256), 0, stream>>>(gn, gnrs, N);

    const float* xs[3] = { feat, feat_a, feat_m };
    float* zf[3] = { out /* hiden_emb region */, zaf, zmf };

    for (int b = 0; b < 3; ++b) {
        conv(xs[b], xb, N, FIN, N, FINP);
        // t1 = x @ w2_1
        gemm(xb, FINP, w21b, HID, t1f, t1b, HID, N, HID, FINP, HID,
             MODE_NONE, 0, 0, 0, 0);
        // z1 = relu(l1*t1 + l2*(adj@t1))
        gemm(adjb, N, t1b, HID, nullptr, z1b, HID, N, HID, N, HID,
             MODE_PROP_RELU, l1, l2, t1f, HID);
        // t2 = z1 @ w2_2
        gemm(z1b, HID, w22b, OUT, t2f, t2b, OUT, N, OUT, HID, OUT,
             MODE_NONE, 0, 0, 0, 0);
        // z = l1*t2 + l2*(adj@t2)
        gemm(adjb, N, t2b, OUT, zf[b], (b == 0 ? zb0 : nullptr), OUT, N, OUT, N, OUT,
             MODE_PROP, l1, l2, t2f, OUT);
        // emb = relu(z) (f32 + bf16 copies)
        relu_conv_k<<<dim3(1024), dim3(256), 0, stream>>>(zf[b], embf[b], embb[b], (long)N * OUT);
        // vsum = graph_neigh @ emb
        gemm(gnb, N, embb[b], OUT, gbuf[b], nullptr, OUT, N, OUT, N, OUT,
             MODE_NONE, 0, 0, 0, 0);
        // g = sigmoid(normalize(vsum / rowsum))
        readout_k<<<dim3(N), dim3(64), 0, stream>>>(gbuf[b], gnrs, OUT);
    }

    // h = adj @ (z @ w2)
    __bf16* yb = xb;  // reuse FINP-wide buffer (padded cols of w2b are zero -> yb pad is zero)
    gemm(zb0, OUT, w2b, FINP, nullptr, yb, FINP, N, FINP, OUT, FINP,
         MODE_NONE, 0, 0, 0, 0);
    // store only the first 3000 columns into d_out (ldc = FIN)
    gemm(adjb, N, yb, FINP, out + off_h, nullptr, FIN, N, FINP, N, FIN,
         MODE_NONE, 0, 0, 0, 0);

    // bilinear discriminators
    disc_k<<<dim3(N), dim3(64), 0, stream>>>(gbuf[0], embf[0], embf[1], dw, db, out + off_ret);
    disc_k<<<dim3(N), dim3(64), 0, stream>>>(gbuf[1], embf[1], embf[0], dw, db, out + off_reta);
    disc_k<<<dim3(N), dim3(64), 0, stream>>>(gbuf[2], embf[2], embf[0], dw, db, out + off_retm);

    // cluster heads: c = z @ cw^T + cb, c_mask = z_mask @ cw^T + cb
    cluster_k<<<dim3((N * NC + 255) / 256), dim3(256), 0, stream>>>(out, cw, cb, out + off_c, N);
    cluster_k<<<dim3((N * NC + 255) / 256), dim3(256), 0, stream>>>(zmf, cw, cb, out + off_cm, N);

    // DINO loss
    hipMemsetAsync(out + off_loss, 0, sizeof(float), stream);
    dino_k<<<dim3((N + 255) / 256), dim3(256), 0, stream>>>(out, zmf, cen, out + off_loss, N);
}
